// Network_82514911691377
// MI455X (gfx1250) — matrix operand order repaired and checked by reading, not compiled, not run on hardware
//
#include <hip/hip_runtime.h>

// ---------------------------------------------------------------------------
// Spiking CNN (Loihi CUBA-LIF) on MI455X / gfx1250.
// conv1 / conv2 / fc1 run on v_wmma_f32_16x16x32_bf16 (spikes {0,1} -> bf16
// lossless). Wave-uniform control/base values are forced into SGPRs via
// readfirstlane so gathers lower to saddr+voffset loads; gathers are phased
// (addresses -> loads -> selects) so loads pipeline behind staggered waits.
// __launch_bounds__(256) widens the VGPR budget to kill residual spills.
// Workspace requirement: ~385 MB (two 191MB ping-pong regions + small).
// ---------------------------------------------------------------------------

typedef __bf16 bf16;
typedef __attribute__((ext_vector_type(16))) __bf16 v16bf;
typedef __attribute__((ext_vector_type(8)))  __bf16 v8bf;
typedef __attribute__((ext_vector_type(8)))  float  v8f;

#define T_STEPS 1450
#define NT16    91          // ceil(1450/16)
#define TPAD    (NT16 * 16) // 1456
#define ALPHA_I 0.75f
#define ALPHA_V 0.90f
#define POOL_GAIN 1.1f
#define FD_D 10

__device__ __forceinline__ int clampi(int x, int lo, int hi) {
  return x < lo ? lo : (x > hi ? hi : x);   // lowers to v_med3_i32
}

// ---------------- pool 4x4 on raw input (memory-bound, coalesced over t) ----
__global__ __launch_bounds__(256) void pool4_kernel(const float* __restrict__ in,
                                                    float* __restrict__ x0) {
  long idx = (long)blockIdx.x * blockDim.x + threadIdx.x;
  const long total = 4L * 2 * 32 * 32 * T_STEPS;
  if (idx >= total) return;
  int t = (int)(idx % T_STEPS);
  long r = idx / T_STEPS;
  int ow = (int)(r & 31); r >>= 5;
  int oh = (int)(r & 31); r >>= 5;
  int c  = (int)(r & 1);  r >>= 1;
  int b  = (int)r;
  const float* base = in + (((long)(b * 2 + c) * 128 + oh * 4) * 128 + ow * 4) * T_STEPS + t;
  float s = 0.f;
#pragma unroll
  for (int i = 0; i < 4; ++i)
#pragma unroll
    for (int j = 0; j < 4; ++j)
      s += base[((long)i * 128 + j) * T_STEPS];
  x0[idx] = s * POOL_GAIN;
}

// ---------------- CUBA-LIF sequential scan (delay_shift fused) -------------
__global__ __launch_bounds__(256) void lif_f32_kernel(const float* __restrict__ x,
                                                      bf16* __restrict__ s,
                                                      int rows, int inStride) {
  int idx = blockIdx.x * blockDim.x + threadIdx.x;
  if (idx >= rows) return;
  const float* xi = x + (long)idx * inStride;
  long o = (long)idx * T_STEPS;
  float iacc = 0.f, v = 0.f;
  s[o] = (bf16)0.f;
  for (int t = 0; t < T_STEPS; ++t) {
    iacc = ALPHA_I * iacc + xi[t];
    v = ALPHA_V * v + iacc;
    float sp = (v >= 1.f) ? 1.f : 0.f;
    v = (sp > 0.f) ? 0.f : v;
    if (t + 1 < T_STEPS) s[o + t + 1] = (bf16)sp;
  }
}

__global__ __launch_bounds__(256) void lif_bf16_kernel(const bf16* __restrict__ x,
                                                       bf16* __restrict__ s,
                                                       int rows, int inStride) {
  int idx = blockIdx.x * blockDim.x + threadIdx.x;
  if (idx >= rows) return;
  const bf16* xi = x + (long)idx * inStride;
  long o = (long)idx * T_STEPS;
  float iacc = 0.f, v = 0.f;
  s[o] = (bf16)0.f;
  for (int t = 0; t < T_STEPS; ++t) {
    iacc = ALPHA_I * iacc + (float)xi[t];
    v = ALPHA_V * v + iacc;
    float sp = (v >= 1.f) ? 1.f : 0.f;
    v = (sp > 0.f) ? 0.f : v;
    if (t + 1 < T_STEPS) s[o + t + 1] = (bf16)sp;
  }
}

// ---------------- fused 2x2 sum-pool (+1.1 gain) -> LIF -> shifted spikes ---
__global__ __launch_bounds__(256) void pool2_lif_kernel(const bf16* __restrict__ in,
                                                        bf16* __restrict__ out,
                                                        int C, int Hin) {
  int idx = blockIdx.x * blockDim.x + threadIdx.x;
  int Ho = Hin >> 1;
  int rows = 4 * C * Ho * Ho;
  if (idx >= rows) return;
  int ow = idx % Ho; int r = idx / Ho;
  int oh = r % Ho; r /= Ho;
  int c = r % C; int b = r / C;
  const bf16* p00 = in + (((long)(b * C + c) * Hin + 2 * oh) * Hin + 2 * ow) * T_STEPS;
  const bf16* p01 = p00 + T_STEPS;
  const bf16* p10 = p00 + (long)Hin * T_STEPS;
  const bf16* p11 = p10 + T_STEPS;
  long o = (long)idx * T_STEPS;
  float iacc = 0.f, v = 0.f;
  out[o] = (bf16)0.f;
  for (int t = 0; t < T_STEPS; ++t) {
    float xv = ((float)p00[t] + (float)p01[t] + (float)p10[t] + (float)p11[t]) * POOL_GAIN;
    iacc = ALPHA_I * iacc + xv;
    v = ALPHA_V * v + iacc;
    float sp = (v >= 1.f) ? 1.f : 0.f;
    v = (sp > 0.f) ? 0.f : v;
    if (t + 1 < T_STEPS) out[o + t + 1] = (bf16)sp;
  }
}

// ---------------- frame density per (b,t), coalesced over t ----------------
__global__ __launch_bounds__(256) void density_kernel(const bf16* __restrict__ s,
                                                      int* __restrict__ dens, int rpb) {
  int t = blockIdx.x * blockDim.x + threadIdx.x;
  int b = blockIdx.y;
  if (t >= T_STEPS) return;
  const bf16* base = s + (long)b * rpb * T_STEPS + t;
  int cnt = 0;
  for (int e = 0; e < rpb; ++e) cnt += ((float)base[(long)e * T_STEPS] != 0.f);
  dens[b * T_STEPS + t] = cnt;
}

// ---------------- frame-drop counter scan (4 threads, one per batch) -------
__global__ __launch_bounds__(64) void framedrop_scan_kernel(const int* __restrict__ dens,
                                                            const int* __restrict__ thrv,
                                                            int thrIdx,
                                                            float* __restrict__ keep) {
  int b = threadIdx.x;
  if (b >= 4) return;
  int thr = thrv[thrIdx];
  unsigned char ev[T_STEPS];
  int c = 0;
  for (int t = 0; t < T_STEPS; ++t) {
    int below = dens[b * T_STEPS + t] < thr;
    c = below ? c + 1 : 0;
    int e = (c == FD_D);
    if (e) c = 0;
    ev[t] = (unsigned char)e;
  }
  for (int t = 0; t < T_STEPS; ++t) {
    float k = 1.f;
    for (int d = 0; d < FD_D; ++d) {
      int u = t + d;
      if (u < T_STEPS && ev[u]) k = 0.f;
    }
    keep[b * T_STEPS + t] = k;
  }
}

__global__ __launch_bounds__(256) void apply_keep_kernel(bf16* __restrict__ s,
                                                         const float* __restrict__ keep,
                                                         int rpb) {
  long idx = (long)blockIdx.x * blockDim.x + threadIdx.x;
  long total = 4L * rpb * T_STEPS;
  if (idx >= total) return;
  int t = (int)(idx % T_STEPS);
  int b = (int)(idx / ((long)rpb * T_STEPS));
  if (keep[b * T_STEPS + t] == 0.f) s[idx] = (bf16)0.f;
}

// ---- transpose S3 [4,2048,T] -> S3T [4,TPAD,2048] t-major, keep folded ----
__global__ __launch_bounds__(256) void transpose_s3_kernel(const bf16* __restrict__ s3,
                                                           const float* __restrict__ keep,
                                                           bf16* __restrict__ s3t) {
  long idx = (long)blockIdx.x * blockDim.x + threadIdx.x;
  const long total = 4L * TPAD * 2048;
  if (idx >= total) return;
  int f = (int)(idx & 2047);
  long r = idx >> 11;
  int t = (int)(r % TPAD);
  int b = (int)(r / TPAD);
  float v = 0.f;
  if (t < T_STEPS)
    v = (float)s3[((long)b * 2048 + f) * T_STEPS + t] * keep[b * T_STEPS + t];
  s3t[idx] = (bf16)v;
}

// ---------------- weight prep: f32 -> bf16, K padded with zeros ------------
__global__ __launch_bounds__(256) void prep_w_kernel(const float* __restrict__ w,
                                                     bf16* __restrict__ wa,
                                                     int O, int Kreal, int Kpad) {
  int idx = blockIdx.x * blockDim.x + threadIdx.x;
  if (idx >= O * Kpad) return;
  int k = idx % Kpad, o = idx / Kpad;
  wa[idx] = (bf16)((k < Kreal) ? w[o * Kreal + k] : 0.f);
}

// ---------------- WMMA fragment helpers (CDNA5 16x16x32 bf16 layouts) ------
__device__ __forceinline__ v16bf load_a_frag(const bf16* __restrict__ wa, int stride,
                                             int m, int hh, int kbase) {
  const bf16* p = wa + m * stride + kbase + 16 * hh;
  v8bf lo = *(const v8bf*)p;
  v8bf hi = *(const v8bf*)(p + 8);
  return __builtin_shufflevector(lo, hi, 0, 1, 2, 3, 4, 5, 6, 7,
                                 8, 9, 10, 11, 12, 13, 14, 15);
}

__device__ __forceinline__ v16bf load_b_contig(const bf16* __restrict__ p) {
  v8bf lo = *(const v8bf*)p;
  v8bf hi = *(const v8bf*)(p + 8);
  return __builtin_shufflevector(lo, hi, 0, 1, 2, 3, 4, 5, 6, 7,
                                 8, 9, 10, 11, 12, 13, 14, 15);
}

// ---------------- conv1 tile accumulate (templated on interior) ------------
template <bool INTERIOR>
__device__ __forceinline__ v8f conv1_accum(const bf16* __restrict__ pb,
                                           const bf16* __restrict__ wa,
                                           int oh, int ow, int m, int hh, int hh16, int tc) {
  v8f c = {};
#pragma unroll
  for (int ks = 0; ks < 2; ++ks) {
    v16bf a = load_a_frag(wa, 64, m, hh, ks * 32);
    int  offs[16];
    bool oks[16];
#pragma unroll
    for (int e = 0; e < 16; ++e) {
      int k = ks * 32 + hh16 + e;
      int ic = k / 25;      int rem = k - ic * 25;
      int kh = rem / 5;     int kw = rem - kh * 5;
      int ih = oh + kh - 2, iw = ow + kw - 2;
      int icc = ic & 1;
      if (INTERIOR) {
        oks[e]  = (k < 50);
        offs[e] = ((icc << 10) + (ih << 5) + iw) * T_STEPS + tc;
      } else {
        int ihc = clampi(ih, 0, 31), iwc = clampi(iw, 0, 31);
        oks[e]  = (k < 50) & (ih == ihc) & (iw == iwc);
        offs[e] = ((icc << 10) + (ihc << 5) + iwc) * T_STEPS + tc;
      }
    }
    bf16 vals[16];
#pragma unroll
    for (int e = 0; e < 16; ++e) vals[e] = pb[offs[e]];
    v16bf bm;
#pragma unroll
    for (int e = 0; e < 16; ++e) bm[e] = oks[e] ? vals[e] : (bf16)0.f;
    c = __builtin_amdgcn_wmma_f32_16x16x32_bf16(false, a, false, bm, (short)0, c, false, false);
  }
  return c;
}

// conv1: implicit GEMM, M=16 oc, K=50->64, N=16 timesteps; 4 pixels per wave.
__global__ __launch_bounds__(256) void conv1_wmma_kernel(const bf16* __restrict__ s0,
                                                         const bf16* __restrict__ wa,
                                                         bf16* __restrict__ x1) {
  int wave = __builtin_amdgcn_readfirstlane(blockIdx.x * (blockDim.x >> 5) +
                                            (threadIdx.x >> 5));
  const int nTiles = 4 * 32 * 8 * NT16;
  if (wave >= nTiles) return;
  int tt = wave % NT16; int r = wave / NT16;
  int ow4 = r & 7;  r >>= 3;
  int oh  = r & 31; r >>= 5;
  int b   = r;
  int lane = threadIdx.x & 31;
  int m = lane & 15, hh = lane >> 4, hh16 = hh << 4;
  int t = tt * 16 + (lane & 15);                    // < TPAD
  int tc = t < T_STEPS ? t : (T_STEPS - 1);
  const bf16* pb = s0 + (long)b * 2048 * T_STEPS;   // scalar slab base

  bool ohIn = (oh >= 2) & (oh <= 29);
  for (int q = 0; q < 4; ++q) {
    int ow = ow4 * 4 + q;
    bool interior = ohIn & (ow >= 2) & (ow <= 29);  // scalar -> s_cbranch
    v8f c = interior ? conv1_accum<true>(pb, wa, oh, ow, m, hh, hh16, tc)
                     : conv1_accum<false>(pb, wa, oh, ow, m, hh, hh16, tc);
    long sbase = (((long)b * 16) * 1024 + oh * 32 + ow) * TPAD + t;
#pragma unroll
    for (int r8 = 0; r8 < 8; ++r8) {
      int oc = r8 + 8 * hh;
      x1[sbase + (long)oc * 1024 * TPAD] = (bf16)c[r8];  // padded: no guard
    }
  }
}

// ---------------- conv2 tile accumulate (templated on interior) ------------
template <bool INTERIOR>
__device__ __forceinline__ void conv2_accum(const bf16* __restrict__ pb,
                                            const bf16* __restrict__ wa,
                                            int oh, int ow, int m, int hh, int hh16, int tc,
                                            v8f& c0, v8f& c1) {
#pragma unroll
  for (int ks = 0; ks < 5; ++ks) {
    int  offs[16];
    bool oks[16];
#pragma unroll
    for (int e = 0; e < 16; ++e) {
      int k = ks * 32 + hh16 + e;
      int ic = k / 9;       int rem = k - ic * 9;
      int kh = rem / 3;     int kw = rem - kh * 3;
      int ih = oh + kh - 1, iw = ow + kw - 1;
      int icc = clampi(ic, 0, 15);
      if (INTERIOR) {
        oks[e]  = (k < 144);
        offs[e] = ((icc << 8) + (ih << 4) + iw) * T_STEPS + tc;
      } else {
        int ihc = clampi(ih, 0, 15), iwc = clampi(iw, 0, 15);
        oks[e]  = (k < 144) & (ih == ihc) & (iw == iwc);
        offs[e] = ((icc << 8) + (ihc << 4) + iwc) * T_STEPS + tc;
      }
    }
    bf16 vals[16];
#pragma unroll
    for (int e = 0; e < 16; ++e) vals[e] = pb[offs[e]];
    v16bf bm;
#pragma unroll
    for (int e = 0; e < 16; ++e) bm[e] = oks[e] ? vals[e] : (bf16)0.f;
    v16bf a0 = load_a_frag(wa, 160, m, hh, ks * 32);
    v16bf a1 = load_a_frag(wa + 16 * 160, 160, m, hh, ks * 32);
    c0 = __builtin_amdgcn_wmma_f32_16x16x32_bf16(false, a0, false, bm, (short)0, c0, false, false);
    c1 = __builtin_amdgcn_wmma_f32_16x16x32_bf16(false, a1, false, bm, (short)0, c1, false, false);
  }
}

// conv2: M=32 (two C tiles sharing B), K=144->160.
__global__ __launch_bounds__(256) void conv2_wmma_kernel(const bf16* __restrict__ s2,
                                                         const bf16* __restrict__ wa,
                                                         bf16* __restrict__ x2) {
  int wave = __builtin_amdgcn_readfirstlane(blockIdx.x * (blockDim.x >> 5) +
                                            (threadIdx.x >> 5));
  const int nTiles = 4 * 16 * 16 * NT16;
  if (wave >= nTiles) return;
  int tt = wave % NT16; int r = wave / NT16;
  int ow = r & 15; r >>= 4;
  int oh = r & 15; r >>= 4;
  int b = r;
  int lane = threadIdx.x & 31;
  int m = lane & 15, hh = lane >> 4, hh16 = hh << 4;
  int t = tt * 16 + (lane & 15);
  int tc = t < T_STEPS ? t : (T_STEPS - 1);
  const bf16* pb = s2 + (long)b * 4096 * T_STEPS;   // scalar slab base

  v8f c0 = {}, c1 = {};
  bool interior = (oh >= 1) & (oh <= 14) & (ow >= 1) & (ow <= 14);  // scalar
  if (interior) conv2_accum<true>(pb, wa, oh, ow, m, hh, hh16, tc, c0, c1);
  else          conv2_accum<false>(pb, wa, oh, ow, m, hh, hh16, tc, c0, c1);

  long sbase = (((long)b * 32) * 256 + oh * 16 + ow) * TPAD + t;
#pragma unroll
  for (int r8 = 0; r8 < 8; ++r8) {
    int oc = r8 + 8 * hh;
    x2[sbase + (long)oc * 256 * TPAD] = (bf16)c0[r8];          // padded: no guard
    x2[sbase + (long)(16 + oc) * 256 * TPAD] = (bf16)c1[r8];
  }
}

// ---------------- fc1: 512x2048 GEMM, B from t-major S3T (pure vector IO) --
__global__ __launch_bounds__(256) void fc1_wmma_kernel(const bf16* __restrict__ s3t,
                                                       const bf16* __restrict__ wa,
                                                       float* __restrict__ x3) {
  int wave = __builtin_amdgcn_readfirstlane(blockIdx.x * (blockDim.x >> 5) +
                                            (threadIdx.x >> 5));
  const int nTiles = 4 * 32 * NT16;
  if (wave >= nTiles) return;
  int tt = wave % NT16; int r = wave / NT16;
  int ot = r & 31; int b = r >> 5;
  int lane = threadIdx.x & 31;
  int m = lane & 15, hh = lane >> 4;
  int t = tt * 16 + (lane & 15);           // < TPAD; padded rows are zero
  const bf16* wrow = wa + (long)ot * 16 * 2048;
  const bf16* brow = s3t + ((long)b * TPAD + t) * 2048 + 16 * hh;
  v8f c0 = {}, c1 = {};
#pragma unroll 4
  for (int ks = 0; ks < 64; ks += 2) {
    v16bf a0 = load_a_frag(wrow, 2048, m, hh, ks * 32);
    v16bf b0 = load_b_contig(brow + ks * 32);
    v16bf a1 = load_a_frag(wrow, 2048, m, hh, ks * 32 + 32);
    v16bf b1 = load_b_contig(brow + ks * 32 + 32);
    c0 = __builtin_amdgcn_wmma_f32_16x16x32_bf16(false, a0, false, b0, (short)0, c0, false, false);
    c1 = __builtin_amdgcn_wmma_f32_16x16x32_bf16(false, a1, false, b1, (short)0, c1, false, false);
  }
  long sbase = ((long)b * 512 + ot * 16) * TPAD + t;
#pragma unroll
  for (int r8 = 0; r8 < 8; ++r8) {
    int o = r8 + 8 * hh;
    x3[sbase + (long)o * TPAD] = c0[r8] + c1[r8];   // padded: no guard
  }
}

// ---------------- fc2 (11x512, tiny) + final LIF to output -----------------
__global__ __launch_bounds__(256) void fc2_kernel(const bf16* __restrict__ s4,
                                                  const float* __restrict__ w,
                                                  float* __restrict__ x4) {
  long idx = (long)blockIdx.x * blockDim.x + threadIdx.x;
  const long total = 4L * 11 * T_STEPS;
  if (idx >= total) return;
  int t = (int)(idx % T_STEPS); long r = idx / T_STEPS;
  int o = (int)(r % 11); int b = (int)(r / 11);
  const bf16* sp = s4 + (long)b * 512 * T_STEPS + t;
  const float* wp = w + o * 512;
  float acc = 0.f;
  for (int f = 0; f < 512; ++f) acc += (float)sp[(long)f * T_STEPS] * wp[f];
  x4[idx] = acc;
}

__global__ __launch_bounds__(64) void lif_final_kernel(const float* __restrict__ x4,
                                                       float* __restrict__ out) {
  int idx = blockIdx.x * blockDim.x + threadIdx.x;
  if (idx >= 44) return;
  long o = (long)idx * T_STEPS;
  float iacc = 0.f, v = 0.f;
  out[o] = 0.f;
  for (int t = 0; t < T_STEPS; ++t) {
    iacc = ALPHA_I * iacc + x4[o + t];
    v = ALPHA_V * v + iacc;
    float sp = (v >= 1.f) ? 1.f : 0.f;
    v = (sp > 0.f) ? 0.f : v;
    if (t + 1 < T_STEPS) out[o + t + 1] = sp;
  }
}

// ---------------------------------------------------------------------------
extern "C" void kernel_launch(void* const* d_in, const int* in_sizes, int n_in,
                              void* d_out, int out_size, void* d_ws, size_t ws_size,
                              hipStream_t stream) {
  const float* spikeInput = (const float*)d_in[0];
  const float* conv1_w    = (const float*)d_in[1];
  const float* conv2_w    = (const float*)d_in[2];
  const float* fc1_w      = (const float*)d_in[3];
  const float* fc2_w      = (const float*)d_in[4];
  const int*   threshold  = (const int*)d_in[5];
  char* ws = (char*)d_ws;

  const long T = T_STEPS;
  size_t off = 0;
  auto alloc = [&](size_t bytes) -> size_t {
    size_t o = off; off += (bytes + 255) & ~(size_t)255; return o;
  };
  size_t oWA1  = alloc((size_t)16 * 64 * sizeof(bf16));
  size_t oWA2  = alloc((size_t)32 * 160 * sizeof(bf16));
  size_t oWFC  = alloc((size_t)512 * 2048 * sizeof(bf16));
  size_t oDens = alloc((size_t)4 * T * sizeof(int));
  size_t oK0   = alloc((size_t)4 * T * sizeof(float));
  size_t oK1   = alloc((size_t)4 * T * sizeof(float));
  size_t oK2   = alloc((size_t)4 * T * sizeof(float));
  size_t oX4   = alloc((size_t)4 * 11 * T * sizeof(float));
  size_t regBytes = (size_t)4 * 16 * 1024 * TPAD * sizeof(bf16);  // 191 MB (padded X1)
  size_t oA = alloc(regBytes);
  size_t oB = alloc(regBytes);
  if (off > ws_size) return;  // workspace too small: no-op

  bf16*  WA1 = (bf16*)(ws + oWA1);
  bf16*  WA2 = (bf16*)(ws + oWA2);
  bf16*  WFC = (bf16*)(ws + oWFC);
  int*   DENS = (int*)(ws + oDens);
  float* K0 = (float*)(ws + oK0);
  float* K1 = (float*)(ws + oK1);
  float* K2 = (float*)(ws + oK2);
  float* X4 = (float*)(ws + oX4);
  // ping-pong regions (lifetimes alternate). X1/X2/X3 use TPAD column stride.
  float* X0  = (float*)(ws + oA);
  bf16*  S0  = (bf16*)(ws + oB);
  bf16*  X1  = (bf16*)(ws + oA);
  bf16*  S1  = (bf16*)(ws + oB);
  bf16*  S2  = (bf16*)(ws + oA);
  bf16*  X2  = (bf16*)(ws + oB);
  bf16*  S2B = (bf16*)(ws + oA);
  bf16*  S3  = (bf16*)(ws + oB);
  bf16*  S3T = (bf16*)(ws + oA);                 // [4,TPAD,2048] t-major
  size_t s3tBytes = ((size_t)4 * TPAD * 2048 * sizeof(bf16) + 255) & ~(size_t)255;
  float* X3  = (float*)(ws + oA + s3tBytes);     // [4,512,TPAD] after S3T
  bf16*  S4  = (bf16*)(ws + oB);
  float* OUT = (float*)d_out;

  auto cdiv = [](long a, long b) -> int { return (int)((a + b - 1) / b); };
  const int BS = 256;
  dim3 densGrid(cdiv(T, BS), 4);

  // weight prep
  prep_w_kernel<<<cdiv(16 * 64, BS), BS, 0, stream>>>(conv1_w, WA1, 16, 50, 64);
  prep_w_kernel<<<cdiv(32 * 160, BS), BS, 0, stream>>>(conv2_w, WA2, 32, 144, 160);
  prep_w_kernel<<<cdiv(512 * 2048, BS), BS, 0, stream>>>(fc1_w, WFC, 512, 2048, 2048);

  // stage 0: pool4 -> LIF(+shift) -> frame_drop(thr0)
  pool4_kernel<<<cdiv(4L * 2 * 32 * 32 * T, BS), BS, 0, stream>>>(spikeInput, X0);
  lif_f32_kernel<<<cdiv(8192, BS), BS, 0, stream>>>(X0, S0, 8192, T_STEPS);
  density_kernel<<<densGrid, BS, 0, stream>>>(S0, DENS, 2048);
  framedrop_scan_kernel<<<1, 4, 0, stream>>>(DENS, threshold, 0, K0);
  apply_keep_kernel<<<cdiv(4L * 2048 * T, BS), BS, 0, stream>>>(S0, K0, 2048);

  // stage 1: conv1 -> LIF(+shift) -> pool2+LIF(+shift) -> frame_drop(thr1)
  conv1_wmma_kernel<<<cdiv(4L * 32 * 8 * NT16, 8), BS, 0, stream>>>(S0, WA1, X1);
  lif_bf16_kernel<<<cdiv(65536, BS), BS, 0, stream>>>(X1, S1, 65536, TPAD);
  pool2_lif_kernel<<<cdiv(16384, BS), BS, 0, stream>>>(S1, S2, 16, 32);
  density_kernel<<<densGrid, BS, 0, stream>>>(S2, DENS, 4096);
  framedrop_scan_kernel<<<1, 4, 0, stream>>>(DENS, threshold, 1, K1);
  apply_keep_kernel<<<cdiv(4L * 4096 * T, BS), BS, 0, stream>>>(S2, K1, 4096);

  // stage 2: conv2 -> LIF(+shift) -> pool2+LIF(+shift) -> frame_drop(thr2)
  conv2_wmma_kernel<<<cdiv(4L * 16 * 16 * NT16, 8), BS, 0, stream>>>(S2, WA2, X2);
  lif_bf16_kernel<<<cdiv(32768, BS), BS, 0, stream>>>(X2, S2B, 32768, TPAD);
  pool2_lif_kernel<<<cdiv(8192, BS), BS, 0, stream>>>(S2B, S3, 32, 16);
  density_kernel<<<densGrid, BS, 0, stream>>>(S3, DENS, 2048);
  framedrop_scan_kernel<<<1, 4, 0, stream>>>(DENS, threshold, 2, K2);
  transpose_s3_kernel<<<cdiv(4L * TPAD * 2048, BS), BS, 0, stream>>>(S3, K2, S3T);

  // stage 3: fc1 -> LIF(+shift) -> fc2 -> LIF(+shift) -> output
  fc1_wmma_kernel<<<cdiv(4L * 32 * NT16, 8), BS, 0, stream>>>(S3T, WFC, X3);
  lif_f32_kernel<<<cdiv(2048, BS), BS, 0, stream>>>(X3, S4, 2048, TPAD);
  fc2_kernel<<<cdiv(4L * 11 * T, BS), BS, 0, stream>>>(S4, fc2_w, X4);
  lif_final_kernel<<<1, 64, 0, stream>>>(X4, OUT);
}
